// Net_two_38431367364977
// MI455X (gfx1250) — compile-verified, matrix operand order checked by reading, not run on hardware
//
#include <hip/hip_runtime.h>
#include <stdint.h>

typedef _Float16 f16;
typedef __attribute__((ext_vector_type(16))) _Float16 v16h;
typedef __attribute__((ext_vector_type(8)))  _Float16 v8h;
typedef __attribute__((ext_vector_type(8)))  float    v8f;

#define BM 128
#define BN 64
#define BK 32
#define LDSA 40   // padded row stride (halves) to spread LDS banks
#define LDSB 40

// ---------------------------------------------------------------------------
// prep kernels
// ---------------------------------------------------------------------------
__global__ void convert_f32_to_f16(const float* __restrict__ in,
                                   f16* __restrict__ out, int n) {
  int i = blockIdx.x * blockDim.x + threadIdx.x;
  if (i < n) out[i] = (f16)in[i];
}

// w: [K][N] fp32  ->  out: [Npad][Kpad] f16 (transposed, zero-padded)
__global__ void transpose_pad_w(const float* __restrict__ w, int K, int N,
                                f16* __restrict__ out, int Kpad, int Npad) {
  int i = blockIdx.x * blockDim.x + threadIdx.x;
  if (i >= Kpad * Npad) return;
  int n = i / Kpad, k = i % Kpad;
  float v = (k < K && n < N) ? w[(size_t)k * N + n] : 0.f;
  out[i] = (f16)v;
}

__global__ void pad_bias(const float* __restrict__ b, int N,
                         float* __restrict__ out, int Npad) {
  int i = blockIdx.x * blockDim.x + threadIdx.x;
  if (i < Npad) out[i] = (i < N) ? b[i] : 0.f;
}

// ---------------------------------------------------------------------------
// async global -> LDS helper (CDNA5 GLOBAL_LOAD_ASYNC_TO_LDS_B128, ASYNCcnt)
// lds addr = low 32 bits of generic shared pointer
// ---------------------------------------------------------------------------
__device__ __forceinline__ void async_copy_b128(const f16* gptr, f16* lptr) {
  unsigned lds = (unsigned)(uintptr_t)lptr;
  unsigned long long ga = (unsigned long long)(uintptr_t)gptr;
  asm volatile("global_load_async_to_lds_b128 %0, %1, off"
               :
               : "v"(lds), "v"(ga)
               : "memory");
}

// ---------------------------------------------------------------------------
// f16 WMMA GEMM:  C[M][ldc] = act( A[M][lda](f16) * Bt[N][ldb]^T (f16) + bias )
// Bt is pre-transposed weights: Bt[n][k]. Block tile 128x64, 8 waves of 32x32.
// Double-buffered LDS with async global->LDS staging overlapped with WMMA.
// All tile dims padded by construction -> no bounds checks.
// ---------------------------------------------------------------------------
__global__ __launch_bounds__(256) void gemm_f16_wmma(
    const f16* __restrict__ A, int lda,
    const f16* __restrict__ Bt, int ldb,
    const float* __restrict__ bias,
    f16* __restrict__ C, int ldc,
    int ktiles, int do_relu) {
  __shared__ f16 sA[2][BM * LDSA];
  __shared__ f16 sB[2][BN * LDSB];

  const int tid  = threadIdx.x;
  const int lane = tid & 31;
  const int wid  = tid >> 5;
  const int wm   = wid >> 1;   // 0..3  (32-row stripe)
  const int wn   = wid & 1;    // 0..1  (32-col stripe)
  const int m0   = blockIdx.y * BM;
  const int n0   = blockIdx.x * BN;

  v8f acc[2][2];
  const v8f vzero = {0.f, 0.f, 0.f, 0.f, 0.f, 0.f, 0.f, 0.f};
#pragma unroll
  for (int i = 0; i < 2; ++i)
#pragma unroll
    for (int j = 0; j < 2; ++j) acc[i][j] = vzero;

  const int mr = lane & 15;          // row/col within 16x16 tile
  const int kh = (lane >> 4) * 8;    // K offset select per ISA A/B layout

  // staging coordinates: 16B chunks; A = rows (tid>>2) and (tid>>2)+64,
  // B = row (tid>>2); seg = (tid&3)*8 halves within the 32-half row.
  const int srow = tid >> 2;
  const int sseg = (tid & 3) * 8;

  // prologue: stage tile 0 into buffer 0 (3 async instructions per wave)
  {
    const f16* a0 = A + (size_t)(m0 + srow) * lda + sseg;
    const f16* a1 = A + (size_t)(m0 + srow + 64) * lda + sseg;
    const f16* b0 = Bt + (size_t)(n0 + srow) * ldb + sseg;
    async_copy_b128(a0, &sA[0][srow * LDSA + sseg]);
    async_copy_b128(a1, &sA[0][(srow + 64) * LDSA + sseg]);
    async_copy_b128(b0, &sB[0][srow * LDSB + sseg]);
  }

  union U { v16h v; v8h h[2]; };

  for (int kt = 0; kt < ktiles; ++kt) {
    const int cur = kt & 1;
    const bool more = (kt + 1) < ktiles;
    if (more) {
      const int nxt = (kt + 1) & 1;
      const int ko = (kt + 1) * BK;
      const f16* a0 = A + (size_t)(m0 + srow) * lda + ko + sseg;
      const f16* a1 = A + (size_t)(m0 + srow + 64) * lda + ko + sseg;
      const f16* b0 = Bt + (size_t)(n0 + srow) * ldb + ko + sseg;
      async_copy_b128(a0, &sA[nxt][srow * LDSA + sseg]);
      async_copy_b128(a1, &sA[nxt][(srow + 64) * LDSA + sseg]);
      async_copy_b128(b0, &sB[nxt][srow * LDSB + sseg]);
      // tile kt resident once only the 3 just-issued ops remain outstanding
      asm volatile("s_wait_asynccnt 0x3" ::: "memory");
    } else {
      asm volatile("s_wait_asynccnt 0x0" ::: "memory");
    }
    __syncthreads();

    // fragments (lanes 0-15: K 0-7 & 16-23 ; lanes 16-31: K 8-15 & 24-31)
    U af[2], bf[2];
#pragma unroll
    for (int i = 0; i < 2; ++i) {
      int r = wm * 32 + i * 16 + mr;
      af[i].h[0] = *reinterpret_cast<const v8h*>(&sA[cur][r * LDSA + kh]);
      af[i].h[1] = *reinterpret_cast<const v8h*>(&sA[cur][r * LDSA + kh + 16]);
    }
#pragma unroll
    for (int j = 0; j < 2; ++j) {
      int r = wn * 32 + j * 16 + mr;
      bf[j].h[0] = *reinterpret_cast<const v8h*>(&sB[cur][r * LDSB + kh]);
      bf[j].h[1] = *reinterpret_cast<const v8h*>(&sB[cur][r * LDSB + kh + 16]);
    }

#pragma unroll
    for (int i = 0; i < 2; ++i)
#pragma unroll
      for (int j = 0; j < 2; ++j)
        acc[i][j] = __builtin_amdgcn_wmma_f32_16x16x32_f16(
            false, af[i].v, false, bf[j].v, (short)0, acc[i][j], false, false);

    // no wave may start overwriting buffer `cur` (next iteration's issue
    // writes tile kt+2 into `cur`,
    // which every wave reaches only after this barrier)
    __syncthreads();
  }

  // epilogue: bias + relu, store f16
  const int hi = (lane >> 4) * 8;
#pragma unroll
  for (int j = 0; j < 2; ++j) {
    int n = n0 + wn * 32 + j * 16 + mr;
    float bv = bias[n];
#pragma unroll
    for (int i = 0; i < 2; ++i) {
#pragma unroll
      for (int r = 0; r < 8; ++r) {
        int m = m0 + wm * 32 + i * 16 + r + hi;
        float v = acc[i][j][r] + bv;
        if (do_relu) v = v > 0.f ? v : 0.f;
        C[(size_t)m * ldc + n] = (f16)v;
      }
    }
  }
}

// ---------------------------------------------------------------------------
// tail: per batch row -> q = h3(128x10) @ w4 + b4, cumsum -> p, then order-4
// log-signature of the 2-channel path. One lane per batch element; the 30-dim
// graded state lives in registers (all inner loops fully unrolled).
// ---------------------------------------------------------------------------
__global__ __launch_bounds__(256) void path_sig_kernel(
    const f16* __restrict__ h3, const float* __restrict__ w4,
    const float* __restrict__ b4, float* __restrict__ p_out,
    float* __restrict__ sig_out) {
  const int bidx = blockIdx.x * blockDim.x + threadIdx.x;

  float W[10][2];
#pragma unroll
  for (int i = 0; i < 10; ++i) { W[i][0] = w4[2 * i]; W[i][1] = w4[2 * i + 1]; }
  const float bb0 = b4[0], bb1 = b4[1];

  const f16* row = h3 + (size_t)bidx * 1280;
  float* pb = p_out + (size_t)bidx * 258;

  float s1[2] = {0.f, 0.f};
  float s2[4] = {0.f};
  float s3[8] = {0.f};
  float s4[16] = {0.f};
  float P0 = 0.f, P1 = 0.f;
  pb[0] = 0.f; pb[1] = 0.f;

  for (int t = 0; t < 128; ++t) {
    float q0 = bb0, q1 = bb1;
#pragma unroll
    for (int i = 0; i < 10; ++i) {
      float hv = (float)row[t * 10 + i];
      q0 += hv * W[i][0];
      q1 += hv * W[i][1];
    }
    P0 += q0; P1 += q1;
    pb[2 + 2 * t] = P0; pb[3 + 2 * t] = P1;

    float d[2] = {q0, q1};
    float c2[4], c3[8], c4[16];
#pragma unroll
    for (int a = 0; a < 2; ++a)
#pragma unroll
      for (int c = 0; c < 2; ++c) c2[a * 2 + c] = d[a] * d[c];
#pragma unroll
    for (int a = 0; a < 4; ++a)
#pragma unroll
      for (int c = 0; c < 2; ++c) c3[a * 2 + c] = c2[a] * d[c];
#pragma unroll
    for (int a = 0; a < 8; ++a)
#pragma unroll
      for (int c = 0; c < 2; ++c) c4[a * 2 + c] = c3[a] * d[c];

    float t2[4], t3[8], t4[16];
#pragma unroll
    for (int i = 0; i < 4; ++i) t2[i] = c2[i] * 0.5f;
#pragma unroll
    for (int i = 0; i < 8; ++i) t3[i] = c3[i] * (1.f / 6.f);
#pragma unroll
    for (int i = 0; i < 16; ++i) t4[i] = c4[i] * (1.f / 24.f);

    // Chen: update high levels first (they read old low levels)
#pragma unroll
    for (int i = 0; i < 16; ++i) s4[i] += t4[i];
#pragma unroll
    for (int a = 0; a < 2; ++a)
#pragma unroll
      for (int c = 0; c < 8; ++c) s4[a * 8 + c] += s1[a] * t3[c];
#pragma unroll
    for (int a = 0; a < 4; ++a)
#pragma unroll
      for (int c = 0; c < 4; ++c) s4[a * 4 + c] += s2[a] * t2[c];
#pragma unroll
    for (int a = 0; a < 8; ++a)
#pragma unroll
      for (int c = 0; c < 2; ++c) s4[a * 2 + c] += s3[a] * d[c];
#pragma unroll
    for (int i = 0; i < 8; ++i) s3[i] += t3[i];
#pragma unroll
    for (int a = 0; a < 2; ++a)
#pragma unroll
      for (int c = 0; c < 4; ++c) s3[a * 4 + c] += s1[a] * t2[c];
#pragma unroll
    for (int a = 0; a < 4; ++a)
#pragma unroll
      for (int c = 0; c < 2; ++c) s3[a * 2 + c] += s2[a] * d[c];
#pragma unroll
    for (int i = 0; i < 4; ++i) s2[i] += t2[i];
#pragma unroll
    for (int a = 0; a < 2; ++a)
#pragma unroll
      for (int c = 0; c < 2; ++c) s2[a * 2 + c] += s1[a] * d[c];
    s1[0] += d[0]; s1[1] += d[1];
  }

  // log(1+S) truncated at order 4 (zero scalar part kills most terms)
  float l2[4], l3[8], l4[16];
#pragma unroll
  for (int i = 0; i < 4; ++i)  l2[i] = s2[i];
#pragma unroll
  for (int i = 0; i < 8; ++i)  l3[i] = s3[i];
#pragma unroll
  for (int i = 0; i < 16; ++i) l4[i] = s4[i];

  float p2_2[4], p2_3[8], p2_4[16];
#pragma unroll
  for (int a = 0; a < 2; ++a)
#pragma unroll
    for (int c = 0; c < 2; ++c) p2_2[a * 2 + c] = s1[a] * s1[c];
#pragma unroll
  for (int i = 0; i < 8; ++i) p2_3[i] = 0.f;
#pragma unroll
  for (int a = 0; a < 2; ++a)
#pragma unroll
    for (int c = 0; c < 4; ++c) p2_3[a * 4 + c] += s1[a] * s2[c];
#pragma unroll
  for (int a = 0; a < 4; ++a)
#pragma unroll
    for (int c = 0; c < 2; ++c) p2_3[a * 2 + c] += s2[a] * s1[c];
#pragma unroll
  for (int i = 0; i < 16; ++i) p2_4[i] = 0.f;
#pragma unroll
  for (int a = 0; a < 2; ++a)
#pragma unroll
    for (int c = 0; c < 8; ++c) p2_4[a * 8 + c] += s1[a] * s3[c];
#pragma unroll
  for (int a = 0; a < 4; ++a)
#pragma unroll
    for (int c = 0; c < 4; ++c) p2_4[a * 4 + c] += s2[a] * s2[c];
#pragma unroll
  for (int a = 0; a < 8; ++a)
#pragma unroll
    for (int c = 0; c < 2; ++c) p2_4[a * 2 + c] += s3[a] * s1[c];
#pragma unroll
  for (int i = 0; i < 4; ++i)  l2[i] -= 0.5f * p2_2[i];
#pragma unroll
  for (int i = 0; i < 8; ++i)  l3[i] -= 0.5f * p2_3[i];
#pragma unroll
  for (int i = 0; i < 16; ++i) l4[i] -= 0.5f * p2_4[i];

  float p3_3[8], p3_4[16];
#pragma unroll
  for (int a = 0; a < 4; ++a)
#pragma unroll
    for (int c = 0; c < 2; ++c) p3_3[a * 2 + c] = p2_2[a] * s1[c];
#pragma unroll
  for (int i = 0; i < 16; ++i) p3_4[i] = 0.f;
#pragma unroll
  for (int a = 0; a < 4; ++a)
#pragma unroll
    for (int c = 0; c < 4; ++c) p3_4[a * 4 + c] += p2_2[a] * s2[c];
#pragma unroll
  for (int a = 0; a < 8; ++a)
#pragma unroll
    for (int c = 0; c < 2; ++c) p3_4[a * 2 + c] += p2_3[a] * s1[c];
#pragma unroll
  for (int i = 0; i < 8; ++i)  l3[i] += (1.f / 3.f) * p3_3[i];
#pragma unroll
  for (int i = 0; i < 16; ++i) l4[i] += (1.f / 3.f) * p3_4[i];

  // P4 level-4 = P3_3 (x) S1
#pragma unroll
  for (int a = 0; a < 8; ++a)
#pragma unroll
    for (int c = 0; c < 2; ++c) l4[a * 2 + c] -= 0.25f * p3_3[a] * s1[c];

  float* sg = sig_out + (size_t)bidx * 30;
  sg[0] = s1[0]; sg[1] = s1[1];
#pragma unroll
  for (int i = 0; i < 4; ++i)  sg[2 + i]  = l2[i];
#pragma unroll
  for (int i = 0; i < 8; ++i)  sg[6 + i]  = l3[i];
#pragma unroll
  for (int i = 0; i < 16; ++i) sg[14 + i] = l4[i];
}

// ---------------------------------------------------------------------------
extern "C" void kernel_launch(void* const* d_in, const int* in_sizes, int n_in,
                              void* d_out, int out_size, void* d_ws,
                              size_t ws_size, hipStream_t stream) {
  (void)in_sizes; (void)n_in; (void)out_size; (void)ws_size;
  const float* x  = (const float*)d_in[0];
  const float* w1 = (const float*)d_in[1];
  const float* b1 = (const float*)d_in[2];
  const float* w2 = (const float*)d_in[3];
  const float* b2 = (const float*)d_in[4];
  const float* w3 = (const float*)d_in[5];
  const float* b3 = (const float*)d_in[6];
  const float* w4 = (const float*)d_in[7];
  const float* b4 = (const float*)d_in[8];

  const int M = 16384;
  char* ws = (char*)d_ws;
  size_t off = 0;
  auto carve = [&](size_t bytes) -> char* {
    char* p = ws + off;
    off = (off + bytes + 255) & ~size_t(255);
    return p;
  };
  f16* xh  = (f16*)carve((size_t)M * 256 * 2);
  f16* h1  = (f16*)carve((size_t)M * 256 * 2);
  f16* h2  = (f16*)carve((size_t)M * 256 * 2);
  f16* h3  = (f16*)carve((size_t)M * 1280 * 2);
  f16* w1T = (f16*)carve((size_t)256 * 256 * 2);
  f16* w2T = (f16*)carve((size_t)256 * 224 * 2);
  f16* w3T = (f16*)carve((size_t)1280 * 224 * 2);
  float* b1p = (float*)carve(256 * 4);
  float* b2p = (float*)carve(256 * 4);

  // prep: f32->f16 convert + weight transpose/pad + bias pad
  convert_f32_to_f16<<<(M * 256 + 255) / 256, 256, 0, stream>>>(x, xh, M * 256);
  transpose_pad_w<<<(256 * 256 + 255) / 256, 256, 0, stream>>>(w1, 256, 200, w1T, 256, 256);
  transpose_pad_w<<<(256 * 224 + 255) / 256, 256, 0, stream>>>(w2, 200, 200, w2T, 224, 256);
  transpose_pad_w<<<(1280 * 224 + 255) / 256, 256, 0, stream>>>(w3, 200, 1280, w3T, 224, 1280);
  pad_bias<<<1, 256, 0, stream>>>(b1, 200, b1p, 256);
  pad_bias<<<1, 256, 0, stream>>>(b2, 200, b2p, 256);

  // GEMM chain (fused bias+relu, f16 activations, f32 accumulate in WMMA)
  gemm_f16_wmma<<<dim3(256 / BN, M / BM), 256, 0, stream>>>(
      xh, 256, w1T, 256, b1p, h1, 256, /*ktiles=*/8, /*relu=*/1);
  gemm_f16_wmma<<<dim3(256 / BN, M / BM), 256, 0, stream>>>(
      h1, 256, w2T, 224, b2p, h2, 256, /*ktiles=*/7, /*relu=*/1);
  gemm_f16_wmma<<<dim3(1280 / BN, M / BM), 256, 0, stream>>>(
      h2, 256, w3T, 224, b3, h3, 1280, /*ktiles=*/7, /*relu=*/1);

  // tail: fc4 + cumsum + order-4 log-signature
  float* p_out = (float*)d_out;                       // (16384, 129, 2)
  float* sig_out = p_out + (size_t)M * 258;           // (16384, 30)
  path_sig_kernel<<<M / 256, 256, 0, stream>>>(h3, w4, b4, p_out, sig_out);
}
